// ContextualLayers_77893526880548
// MI455X (gfx1250) — compile-verified, matrix operand order checked by reading, not run on hardware
//
#include <hip/hip_runtime.h>
#include <hip/hip_bf16.h>

typedef __attribute__((ext_vector_type(16))) __bf16 v16bf;
typedef __attribute__((ext_vector_type(8)))  __bf16 v8bf;
typedef __attribute__((ext_vector_type(8)))  float  v8f;

static constexpr int kN  = 100000;   // nodes (divisible by 16)
static constexpr int kE  = 1600000;  // edges
static constexpr int kIN = 128;      // in feats
static constexpr int kH  = 64;       // hidden feats
static constexpr float kSlope = 0.2f;

// ---------------------------------------------------------------------------
// GEMM: feat[N,64] = X[N,Din] @ W[64,Din]^T   (bf16 WMMA, fp32 accumulate)
// W is converted to bf16 once per block into padded LDS; B fragments come
// from LDS via ds_load_b128. One wave -> 16 rows x 64 cols (4 WMMA tiles).
// ---------------------------------------------------------------------------
template<int DIN>
__global__ __launch_bounds__(128) void gat_gemm_kernel(
    const float* __restrict__ X, const float* __restrict__ W,
    float* __restrict__ feat)
{
  constexpr int LDW = DIN + 8;            // padded row stride (bf16 elems)
  __shared__ __bf16 lw[64 * LDW];

  const int tid = threadIdx.x;
  // cooperative fp32 -> bf16 conversion of W into LDS (once per block)
  for (int idx = tid; idx < 64 * DIN; idx += 128) {
    const int n = idx / DIN, k = idx % DIN;
    lw[n * LDW + k] = (__bf16)W[idx];
  }
  __syncthreads();

  const int lane = tid & 31;
  const int wave = tid >> 5;
  const int tile = blockIdx.x * 4 + wave;         // 16-row tile index
  if (tile * 16 >= kN) return;                    // wave-uniform exit

  const int mrow = tile * 16 + (lane & 15);       // A row for this lane
  const int kph  = (lane < 16) ? 0 : 8;           // K phase per ISA A-layout
  const int wn   = lane & 15;                     // B column within tile

  v8f acc[4] = {v8f{}, v8f{}, v8f{}, v8f{}};

  const float* __restrict__ xrow = X + (size_t)mrow * DIN;

  for (int kb = 0; kb < DIN; kb += 32) {
    // A fragment: 16 bf16 per lane: K = kb+kph..+7 and kb+16+kph..+7
    v16bf va;
#pragma unroll
    for (int j = 0; j < 8; ++j) {
      va[j]     = (__bf16)xrow[kb + kph + j];
      va[j + 8] = (__bf16)xrow[kb + 16 + kph + j];
    }
#pragma unroll
    for (int nt = 0; nt < 4; ++nt) {
      const __bf16* wr = lw + (size_t)(nt * 16 + wn) * LDW + kb + kph;
      const v8bf lo = *(const v8bf*)wr;          // ds_load_b128
      const v8bf hi = *(const v8bf*)(wr + 16);   // ds_load_b128
      const v16bf vb = __builtin_shufflevector(
          lo, hi, 0, 1, 2, 3, 4, 5, 6, 7, 8, 9, 10, 11, 12, 13, 14, 15);
      acc[nt] = __builtin_amdgcn_wmma_f32_16x16x32_bf16(
          false, va, false, vb, (short)0, acc[nt], false, false);
    }
  }

  // D layout: vgpr v -> (M = v or v+8, N = lane&15)
  const int moff = (lane < 16) ? 0 : 8;
  float* __restrict__ out = feat + (size_t)(tile * 16) * kH;
#pragma unroll
  for (int v = 0; v < 8; ++v) {
    const size_t r = (size_t)(v + moff) * kH;
#pragma unroll
    for (int nt = 0; nt < 4; ++nt)
      out[r + nt * 16 + wn] = acc[nt][v];
  }
}

// ---------------------------------------------------------------------------
// el[i] = feat[i,:].al ; er[i] = feat[i,:].ar
// ---------------------------------------------------------------------------
__global__ void elr_kernel(const float* __restrict__ feat,
                           const float* __restrict__ al,
                           const float* __restrict__ ar,
                           float* __restrict__ el, float* __restrict__ er)
{
  const int i = blockIdx.x * blockDim.x + threadIdx.x;
  if (i >= kN) return;
  const float4* f = (const float4*)(feat + (size_t)i * kH);
  const float4* A = (const float4*)al;
  const float4* B = (const float4*)ar;
  float sl = 0.f, sr = 0.f;
#pragma unroll 4
  for (int q = 0; q < kH / 4; ++q) {
    float4 v = f[q], a = A[q], b = B[q];
    sl += v.x * a.x + v.y * a.y + v.z * a.z + v.w * a.w;
    sr += v.x * b.x + v.y * b.y + v.z * b.z + v.w * b.w;
  }
  el[i] = sl;
  er[i] = sr;
}

// ordered-uint encoding so atomicMax(uint) implements float max
__device__ __forceinline__ unsigned f2ord(float f) {
  unsigned u = __float_as_uint(f);
  return (u & 0x80000000u) ? ~u : (u | 0x80000000u);
}
__device__ __forceinline__ float ord2f(unsigned u) {
  return (u & 0x80000000u) ? __uint_as_float(u & 0x7FFFFFFFu)
                           : __uint_as_float(~u);
}

// e = leaky_relu(el[src]+er[dst]); segment-max into mord[dst]
__global__ void edge_logit_kernel(const float* __restrict__ el,
                                  const float* __restrict__ er,
                                  const int* __restrict__ src,
                                  const int* __restrict__ dst,
                                  float* __restrict__ ev,
                                  unsigned* __restrict__ mord)
{
  const int i = blockIdx.x * blockDim.x + threadIdx.x;
  if (i >= kE) return;
  float t = el[src[i]] + er[dst[i]];
  t = (t > 0.f) ? t : kSlope * t;
  ev[i] = t;
  atomicMax(&mord[dst[i]], f2ord(t));
}

// ee = exp(e - m[dst]); segment-sum into denom[dst]
__global__ void edge_exp_kernel(float* __restrict__ ev,
                                const unsigned* __restrict__ mord,
                                const int* __restrict__ dst,
                                float* __restrict__ denom)
{
  const int i = blockIdx.x * blockDim.x + threadIdx.x;
  if (i >= kE) return;
  const float m  = ord2f(mord[dst[i]]);
  const float ee = __expf(ev[i] - m);
  ev[i] = ee;
  atomicAdd(&denom[dst[i]], ee);
}

// agg[dst] += feat[src] * (ee/denom[dst]); 16 threads per edge, float4 each
__global__ void edge_agg_kernel(const float* __restrict__ ev,
                                const float* __restrict__ denom,
                                const int* __restrict__ src,
                                const int* __restrict__ dst,
                                const float* __restrict__ feat,
                                float* __restrict__ agg)
{
  const int idx = blockIdx.x * blockDim.x + threadIdx.x;
  if (idx >= kE * 16) return;
  const int ei = idx >> 4;
  const int hq = (idx & 15) << 2;
  const int s = src[ei], d = dst[ei];
  const float alpha = ev[ei] / denom[d];
  const float4 v = *(const float4*)(feat + (size_t)s * kH + hq);
  float* o = agg + (size_t)d * kH + hq;
  atomicAdd(o + 0, v.x * alpha);
  atomicAdd(o + 1, v.y * alpha);
  atomicAdd(o + 2, v.z * alpha);
  atomicAdd(o + 3, v.w * alpha);
}

// out = relu(agg + b)
__global__ void bias_relu_kernel(const float* __restrict__ agg,
                                 const float* __restrict__ b,
                                 float* __restrict__ out)
{
  const int i = blockIdx.x * blockDim.x + threadIdx.x;
  if (i >= kN * kH) return;
  const float v = agg[i] + b[i & (kH - 1)];
  out[i] = (v > 0.f) ? v : 0.f;
}

// ---------------------------------------------------------------------------
static void run_layer(const float* x, int din, const float* W, const float* al,
                      const float* ar, const float* b, const int* src,
                      const int* dst, float* feat, float* el, float* er,
                      unsigned* mord, float* denom, float* ev, float* agg,
                      float* out_h, hipStream_t stream)
{
  hipMemsetAsync(agg,   0, (size_t)kN * kH * sizeof(float), stream);
  hipMemsetAsync(mord,  0, (size_t)kN * sizeof(unsigned),   stream);
  hipMemsetAsync(denom, 0, (size_t)kN * sizeof(float),      stream);

  const int tiles = kN / 16;                 // 6250
  const int gemm_blocks = (tiles + 3) / 4;   // 4 waves (tiles) per block
  if (din == kIN)
    gat_gemm_kernel<kIN><<<gemm_blocks, 128, 0, stream>>>(x, W, feat);
  else
    gat_gemm_kernel<kH><<<gemm_blocks, 128, 0, stream>>>(x, W, feat);

  elr_kernel<<<(kN + 255) / 256, 256, 0, stream>>>(feat, al, ar, el, er);
  edge_logit_kernel<<<(kE + 255) / 256, 256, 0, stream>>>(el, er, src, dst, ev, mord);
  edge_exp_kernel<<<(kE + 255) / 256, 256, 0, stream>>>(ev, mord, dst, denom);
  edge_agg_kernel<<<(kE * 16 + 255) / 256, 256, 0, stream>>>(ev, denom, src, dst, feat, agg);
  bias_relu_kernel<<<(kN * kH + 255) / 256, 256, 0, stream>>>(agg, b, out_h);
}

extern "C" void kernel_launch(void* const* d_in, const int* in_sizes, int n_in,
                              void* d_out, int out_size, void* d_ws, size_t ws_size,
                              hipStream_t stream)
{
  const float* in_feat = (const float*)d_in[0];
  const float* W1  = (const float*)d_in[1];
  const float* al1 = (const float*)d_in[2];
  const float* ar1 = (const float*)d_in[3];
  const float* b1  = (const float*)d_in[4];
  const float* W2  = (const float*)d_in[5];
  const float* al2 = (const float*)d_in[6];
  const float* ar2 = (const float*)d_in[7];
  const float* b2  = (const float*)d_in[8];
  const float* W3  = (const float*)d_in[9];
  const float* al3 = (const float*)d_in[10];
  const float* ar3 = (const float*)d_in[11];
  const float* b3  = (const float*)d_in[12];
  const int*   src = (const int*)d_in[13];
  const int*   dst = (const int*)d_in[14];

  // workspace carve-out (~86 MB)
  float*    feat  = (float*)d_ws;                       // N*64
  float*    hbuf  = feat + (size_t)kN * kH;             // N*64
  float*    agg   = hbuf + (size_t)kN * kH;             // N*64
  float*    el    = agg  + (size_t)kN * kH;             // N
  float*    er    = el + kN;                            // N
  unsigned* mord  = (unsigned*)(er + kN);               // N
  float*    denom = (float*)(mord + kN);                // N
  float*    ev    = denom + kN;                         // E

  float* out = (float*)d_out;

  run_layer(in_feat, kIN, W1, al1, ar1, b1, src, dst,
            feat, el, er, mord, denom, ev, agg, hbuf, stream);
  run_layer(hbuf, kH, W2, al2, ar2, b2, src, dst,
            feat, el, er, mord, denom, ev, agg, hbuf, stream);
  run_layer(hbuf, kH, W3, al3, ar3, b3, src, dst,
            feat, el, er, mord, denom, ev, agg, out, stream);
}
